// MoEExpertRouter_66099546685646
// MI455X (gfx1250) — compile-verified, matrix operand order checked by reading, not run on hardware
//
#include <hip/hip_runtime.h>
#include <hip/hip_bf16.h>
#include <stdint.h>

#define BATCH 4
#define SEQ   2048
#define HID   8192
#define NEXP  256
#define TOPK  8
#define MROWS (BATCH * SEQ)   // 8192 token rows
#define MTILE 32              // rows per workgroup
#define KTILE 32              // K per pipeline stage (one bf16 WMMA K)

typedef __attribute__((ext_vector_type(16))) __bf16      v16bf;
typedef __attribute__((ext_vector_type(8)))  float        v8f;
typedef __attribute__((ext_vector_type(4)))  unsigned int v4u;
typedef __attribute__((ext_vector_type(8)))  int          v8i;
typedef __attribute__((ext_vector_type(4)))  int          v4i;

// Pun bf16 fragment as scalars for element fill; hardware cvt does RNE.
union FragBF { v16bf v; __bf16 h[16]; };

#ifndef __has_builtin
#define __has_builtin(x) 0
#endif
#if __has_builtin(__builtin_amdgcn_tensor_load_to_lds) && \
    __has_builtin(__builtin_amdgcn_s_wait_tensorcnt)
#define USE_TDM 1
#else
#define USE_TDM 0
#endif

#if USE_TDM
// Build D# groups for a 2D f32 tile load (tile fully in-bounds) and fire TDM.
// tensor/tile dims and stride are in 4-byte elements.
// Toolchain builtin arity is 6: (v4u g0, v8i g1, v4i g2, v4i g3, v8i g4, i32 cpol)
__device__ __forceinline__ void tdm_load_2d(unsigned lds_off, const void* gptr,
                                            unsigned tensor_d0, unsigned tensor_d1,
                                            unsigned tile_d0, unsigned tile_d1,
                                            unsigned long long stride0) {
  unsigned long long ga = (unsigned long long)gptr;
  v4u g0;
  g0.x = 1u;                                          // count=1, user mode
  g0.y = lds_off;                                     // lds_addr (bytes)
  g0.z = (unsigned)(ga & 0xffffffffu);                // global_addr[31:0]
  g0.w = (unsigned)((ga >> 32) & 0x1ffffffu)          // global_addr[56:32]
       | (2u << 30);                                  // type=2 ("image")
  v8i g1;
  g1[0] = (int)(2u << 16);                            // data_size=4B, mask=0
  g1[1] = (int)((tensor_d0 & 0xffffu) << 16);         // dim0[15:0]
  g1[2] = (int)((tensor_d0 >> 16) | ((tensor_d1 & 0xffffu) << 16));
  g1[3] = (int)((tensor_d1 >> 16) | (tile_d0 << 16)); // dim1 hi | tile_dim0
  g1[4] = (int)(tile_d1 & 0xffffu);                   // tile_dim1 (tile_dim2=0)
  g1[5] = (int)(stride0 & 0xffffffffu);               // dim0_stride[31:0]
  g1[6] = (int)((stride0 >> 32) & 0xffffu);           // dim0_stride[47:32]
  g1[7] = 0;
  v4i z4 = {0, 0, 0, 0};
  v8i z8 = {0, 0, 0, 0, 0, 0, 0, 0};
  __builtin_amdgcn_tensor_load_to_lds(g0, g1, z4, z4, z8, 0);
}
#endif

// One K=32 step. Build A frag and ALL FOUR B frags in distinct registers
// first, then issue the 4 WMMAs back-to-back (no source-reg overlap between
// consecutive WMMAs -> no WAR hazard v_nops inside the group; only one 4-slot
// hazard window per step, fillable with next step's ds_loads).
// Frag element order follows ISA 16-bit layouts: lanes 0-15 hold
// K {0..7,16..23}, lanes 16-31 hold K {8..15,24..31}.
__device__ __forceinline__ void mma_step(const float* __restrict__ As,
                                         const float* __restrict__ Bs,
                                         int lane, int rowhalf, int colbase,
                                         v8f acc[4]) {
  const int m  = lane & 15;
  const int lo = (lane >> 4) << 3;            // 0 or 8
  FragBF a;
  {
    const float*  ar = As + (rowhalf * 16 + m) * KTILE;
    const float4* p0 = (const float4*)(ar + lo);
    const float4* p1 = (const float4*)(ar + 16 + lo);
    float4 f0 = p0[0], f1 = p0[1], f2 = p1[0], f3 = p1[1];
    a.h[0]  = (__bf16)f0.x; a.h[1]  = (__bf16)f0.y;
    a.h[2]  = (__bf16)f0.z; a.h[3]  = (__bf16)f0.w;
    a.h[4]  = (__bf16)f1.x; a.h[5]  = (__bf16)f1.y;
    a.h[6]  = (__bf16)f1.z; a.h[7]  = (__bf16)f1.w;
    a.h[8]  = (__bf16)f2.x; a.h[9]  = (__bf16)f2.y;
    a.h[10] = (__bf16)f2.z; a.h[11] = (__bf16)f2.w;
    a.h[12] = (__bf16)f3.x; a.h[13] = (__bf16)f3.y;
    a.h[14] = (__bf16)f3.z; a.h[15] = (__bf16)f3.w;
  }
  FragBF b[4];
#pragma unroll
  for (int t = 0; t < 4; ++t) {
    const int c = colbase + t * 16 + m;
#pragma unroll
    for (int p = 0; p < 4; ++p) {
      b[t].h[2 * p]         = (__bf16)Bs[(lo + 2 * p)          * NEXP + c];
      b[t].h[2 * p + 1]     = (__bf16)Bs[(lo + 2 * p + 1)      * NEXP + c];
      b[t].h[8 + 2 * p]     = (__bf16)Bs[(16 + lo + 2 * p)     * NEXP + c];
      b[t].h[8 + 2 * p + 1] = (__bf16)Bs[(16 + lo + 2 * p + 1) * NEXP + c];
    }
  }
#pragma unroll
  for (int t = 0; t < 4; ++t) {
    acc[t] = __builtin_amdgcn_wmma_f32_16x16x32_bf16(
        false, a.v, false, b[t].v, (short)0, acc[t], false, false);
  }
}

__global__ void __launch_bounds__(256)
moe_router_fused(const float* __restrict__ x, const float* __restrict__ W,
                 float* __restrict__ out_w, int* __restrict__ out_i,
                 float* __restrict__ out_logits) {
  extern __shared__ float smem[];
  float* A0 = smem;           // 1024 f32 (4 KB)
  float* A1 = smem + 1024;    // 1024 f32
  float* B0 = smem + 2048;    // 8192 f32 (32 KB)
  float* B1 = smem + 10240;   // 8192 f32
  float* Ls = B0;             // 32x256 logits tile, aliases B0 after the loop

  const int tid     = (int)threadIdx.x;
  const int lane    = tid & 31;
  const int wave    = tid >> 5;
  const int rowhalf = wave >> 2;          // waves 0-3: rows 0-15, 4-7: rows 16-31
  const int colbase = (wave & 3) * 64;    // 64 experts per wave
  const long long row0 = (long long)blockIdx.x * MTILE;

  v8f zero = {};
  v8f acc[4];
#pragma unroll
  for (int t = 0; t < 4; ++t) acc[t] = zero;

  const int NK = HID / KTILE;  // 256 stages

#if USE_TDM
  if (wave == 0) {
    tdm_load_2d((unsigned)(uintptr_t)A0, x + row0 * HID,
                HID, MTILE, KTILE, MTILE, (unsigned long long)HID);
    tdm_load_2d((unsigned)(uintptr_t)B0, W,
                NEXP, HID, NEXP, KTILE, (unsigned long long)NEXP);
  }
  for (int kb = 0; kb < NK; ++kb) {
    float* Ab = (kb & 1) ? A1 : A0;
    float* Bb = (kb & 1) ? B1 : B0;
    if (wave == 0) __builtin_amdgcn_s_wait_tensorcnt(0);
    __syncthreads();  // publishes current tiles AND fences last reads of other buf
    if (wave == 0 && kb + 1 < NK) {
      float* An = (kb & 1) ? A0 : A1;
      float* Bn = (kb & 1) ? B0 : B1;
      const long long kn = (long long)(kb + 1) * KTILE;
      tdm_load_2d((unsigned)(uintptr_t)An, x + row0 * HID + kn,
                  HID, MTILE, KTILE, MTILE, (unsigned long long)HID);
      tdm_load_2d((unsigned)(uintptr_t)Bn, W + kn * NEXP,
                  NEXP, HID, NEXP, KTILE, (unsigned long long)NEXP);
    }
    mma_step(Ab, Bb, lane, rowhalf, colbase, acc);
  }
#else
  for (int kb = 0; kb < NK; ++kb) {
    {  // A tile: 256 float4, one per thread
      const int r = tid >> 3, q = tid & 7;
      ((float4*)A0)[tid] =
          *(const float4*)(x + (row0 + r) * HID + (long long)kb * KTILE + q * 4);
    }
#pragma unroll
    for (int i = 0; i < 8; ++i) {  // B tile: 2048 float4
      ((float4*)B0)[tid + i * 256] =
          ((const float4*)(W + (long long)kb * KTILE * NEXP))[tid + i * 256];
    }
    __syncthreads();
    mma_step(A0, B0, lane, rowhalf, colbase, acc);
    __syncthreads();
  }
#endif

  // ---- Epilogue: C tiles -> LDS (ISA C-layout: lanes 0-15 M=r, 16-31 M=r+8)
#pragma unroll
  for (int t = 0; t < 4; ++t) {
    const int m0 = rowhalf * 16 + ((lane >> 4) << 3);
    const int n  = colbase + t * 16 + (lane & 15);
#pragma unroll
    for (int r = 0; r < 8; ++r) Ls[(m0 + r) * NEXP + n] = acc[t][r];
  }
  __syncthreads();

  // Stream logits out coalesced (thread = expert column)
#pragma unroll 4
  for (int i = 0; i < MTILE; ++i)
    out_logits[(row0 + i) * NEXP + tid] = Ls[i * NEXP + tid];

  // Top-8 + softmax: one wave per row, 4 rows per wave
  for (int rr = 0; rr < 4; ++rr) {
    const int r = wave * 4 + rr;
    float v[8];
    {
      const float4* p = (const float4*)(Ls + r * NEXP + lane * 8);
      float4 q0 = p[0], q1 = p[1];
      v[0] = q0.x; v[1] = q0.y; v[2] = q0.z; v[3] = q0.w;
      v[4] = q1.x; v[5] = q1.y; v[6] = q1.z; v[7] = q1.w;
    }
    float topv[TOPK]; int topi[TOPK];
#pragma unroll
    for (int t = 0; t < TOPK; ++t) {
      float bm = -3.4e38f; int bi = 0x7fffffff;
#pragma unroll
      for (int j = 0; j < 8; ++j)
        if (v[j] > bm) { bm = v[j]; bi = lane * 8 + j; }
#pragma unroll
      for (int off = 16; off > 0; off >>= 1) {  // wave32 argmax, low-index ties
        float om = __shfl_xor(bm, off);
        int   oi = __shfl_xor(bi, off);
        if (om > bm || (om == bm && oi < bi)) { bm = om; bi = oi; }
      }
      topv[t] = bm; topi[t] = bi;
      const int loc = bi - lane * 8;
      if (loc >= 0 && loc < 8) v[loc] = -3.4e38f;
    }
    float s = 0.f;
#pragma unroll
    for (int t = 0; t < TOPK; ++t) s += __expf(topv[t] - topv[0]);
    if (lane < TOPK) {
      const long long g = row0 + r;
      out_w[g * TOPK + lane] = __expf(topv[lane] - topv[0]) / s;
      out_i[g * TOPK + lane] = topi[lane];
    }
  }
}

extern "C" void kernel_launch(void* const* d_in, const int* in_sizes, int n_in,
                              void* d_out, int out_size, void* d_ws, size_t ws_size,
                              hipStream_t stream) {
  (void)in_sizes; (void)n_in; (void)out_size; (void)d_ws; (void)ws_size;
  const float* x = (const float*)d_in[0];
  const float* W = (const float*)d_in[1];
  float* out_w = (float*)d_out;                                  // [8192,8] f32
  int*   out_i = (int*)d_out + (size_t)MROWS * TOPK;             // [8192,8] i32
  float* out_l = (float*)d_out + (size_t)2 * MROWS * TOPK;       // [8192,256] f32
  dim3 grid(MROWS / MTILE);   // 256 workgroups
  dim3 block(256);            // 8 waves (wave32)
  size_t shmem = (size_t)18432 * sizeof(float);  // 72 KB double-buffered tiles
  moe_router_fused<<<grid, block, shmem, stream>>>(x, W, out_w, out_i, out_l);
}
